// Attention_47811575939181
// MI455X (gfx1250) — compile-verified, matrix operand order checked by reading, not run on hardware
//
#include <hip/hip_runtime.h>
#include <hip/hip_bf16.h>

#define B_   16
#define N_   1024
#define DIM_ 512
#define H_   8
#define DH_  64
#define HD_  512   // H*DH

typedef __attribute__((ext_vector_type(16))) __bf16 v16bf;
typedef __attribute__((ext_vector_type(8)))  float  v8f;

union Frag { v16bf v; uint4 u[2]; unsigned short s[16]; };
union Acc  { v8f v; float f[8]; };

__device__ __forceinline__ unsigned short f2bf(float f) {
  unsigned u = __float_as_uint(f);
  u += 0x7FFFu + ((u >> 16) & 1u);          // round-to-nearest-even
  return (unsigned short)(u >> 16);
}
__device__ __forceinline__ float bf2f(unsigned short h) {
  return __uint_as_float(((unsigned)h) << 16);
}

// 16-bf16 WMMA A/B fragment from a row-major bf16 matrix where this row's
// (or, for B-from-transposed, this column's) 32-element K-run is contiguous.
// Layout (ISA 7.12.2, 16-bit A 16x32): lanes 0-15 hold K {0..7,16..23},
// lanes 16-31 hold K {8..15,24..31}; B mirrors with N=lane.
__device__ __forceinline__ v16bf load_frag(const unsigned short* p, int ksel) {
  Frag f;
  const uint4* q = reinterpret_cast<const uint4*>(p + ksel * 8);
  f.u[0] = q[0];   // K-run elems 0..7
  f.u[1] = q[2];   // K-run elems 16..23 (+16 bf16 = +2 uint4)
  return f.v;
}

// ---------------------------------------------------------------- conversions
__global__ void k_cvt(const float* __restrict__ src, unsigned short* __restrict__ dst, int n) {
  int i = blockIdx.x * blockDim.x + threadIdx.x;
  if (i < n) dst[i] = f2bf(src[i]);
}

// src [rows][cols] fp32 row-major  ->  dst [cols][rows] bf16 (transposed)
__global__ void k_cvtT(const float* __restrict__ src, unsigned short* __restrict__ dst,
                       int rows, int cols) {
  int i = blockIdx.x * blockDim.x + threadIdx.x;
  if (i < rows * cols) {
    int r = i / cols, c = i % cols;
    dst[c * rows + r] = f2bf(src[i]);
  }
}

// ------------------------------------------------------- QKV projection GEMM
// grid (B*N/16, 24): ct<8 -> Q (8 x 64 cols), ct>=8 -> KV (16 x 64 cols).
// One wave per block computes a 16x64 tile: 4 wmma share one A fragment.
__global__ void __launch_bounds__(32)
k_proj(const unsigned short* __restrict__ xb,
       const unsigned short* __restrict__ wqT,
       const unsigned short* __restrict__ wkvT,
       unsigned short* __restrict__ qb,
       unsigned short* __restrict__ kb,
       unsigned short* __restrict__ vTb) {
  const int lane = threadIdx.x & 31;
  const int ksel = lane >> 4;
  const int mt   = blockIdx.x;
  const int ct   = blockIdx.y;
  const bool isQ = ct < 8;
  const int obase = (isQ ? ct : (ct - 8)) * 64;
  const unsigned short* wT = isQ ? wqT : wkvT;

  const unsigned short* arow  = xb + (mt * 16 + (lane & 15)) * DIM_;
  const unsigned short* brow0 = wT + (obase + (lane & 15)) * DIM_;

  Acc acc[4];
#pragma unroll
  for (int j = 0; j < 4; ++j)
#pragma unroll
    for (int i = 0; i < 8; ++i) acc[j].f[i] = 0.f;

  for (int k = 0; k < DIM_; k += 32) {
    __builtin_prefetch(arow + k + 64, 0, 1);   // global_prefetch_b8
    v16bf a = load_frag(arow + k, ksel);
#pragma unroll
    for (int j = 0; j < 4; ++j) {
      __builtin_prefetch(brow0 + j * 16 * DIM_ + k + 64, 0, 1);
      v16bf b = load_frag(brow0 + j * 16 * DIM_ + k, ksel);
      acc[j].v = __builtin_amdgcn_wmma_f32_16x16x32_bf16(false, a, false, b,
                                                         (short)0, acc[j].v,
                                                         false, false);
    }
  }

#pragma unroll
  for (int j = 0; j < 4; ++j) {
    const int o = obase + j * 16 + (lane & 15);
#pragma unroll
    for (int v = 0; v < 8; ++v) {
      int mrow = mt * 16 + v + 8 * ksel;       // C layout: M = v + 8*(lane>>4)
      int bb = mrow >> 10, n = mrow & (N_ - 1);
      unsigned short val = f2bf(acc[j].f[v]);
      if (isQ) {
        int h = o >> 6, d = o & 63;
        qb[((bb * H_ + h) * N_ + n) * DH_ + d] = val;
      } else if (o < HD_) {                    // kv[...,0,...] = K
        int h = o >> 6, d = o & 63;
        kb[((bb * H_ + h) * N_ + n) * DH_ + d] = val;
      } else {                                 // kv[...,1,...] = V, transposed
        int o2 = o - HD_;
        int h = o2 >> 6, d = o2 & 63;
        vTb[((bb * H_ + h) * DH_ + d) * N_ + n] = val;
      }
    }
  }
}

// ------------------------------------------------- fused talking-heads attn
// One block per (batch, 16-query-row tile); 8 waves, wave == head.
// Dynamic LDS: S[8][16][1024] bf16 (256KB) + RAW[8][16][128] bf16 (32KB)
//            + INVL[8][16] + PRE[8][8] + POST[8][8] + SCL[8] floats (~290KB
//            of the 320KB/WGP — impossible on CDNA4, the point of CDNA5 LDS).
__global__ void __launch_bounds__(256)
k_attn(const unsigned short* __restrict__ qb,
       const unsigned short* __restrict__ kb,
       const unsigned short* __restrict__ vTb,
       const float* __restrict__ scale,
       const float* __restrict__ mix_pre,
       const float* __restrict__ mix_post,
       unsigned short* __restrict__ ab) {
  extern __shared__ __align__(16) char smem[];
  unsigned short* S   = (unsigned short*)smem;          // [8][16][1024]
  unsigned short* RAW = S + 8 * 16 * 1024;              // [8][16][128]
  float* INVL = (float*)(RAW + 8 * 16 * 128);           // [8][16]
  float* PRE  = INVL + 8 * 16;                          // [8][8]
  float* POST = PRE + 64;                               // [8][8]
  float* SCL  = POST + 64;                              // [8]

  const int tid  = threadIdx.x;
  const int lane = tid & 31;
  const int wave = tid >> 5;       // head index
  const int ksel = lane >> 4;
  const int mrow = lane & 15;

  const int blk = blockIdx.x;      // 0..B*N/16-1
  const int bb  = blk >> 6;
  const int i0  = (blk & 63) * 16;

  if (tid < 64) { PRE[tid] = mix_pre[tid]; POST[tid] = mix_post[tid]; }
  if (tid < 8)  SCL[tid] = scale[tid];
  __syncthreads();

  // Q fragments for this head (full K = DH = 64), resident in VGPRs
  const unsigned short* qrow = qb + ((bb * H_ + wave) * N_ + (i0 + mrow)) * DH_;
  v16bf aq0 = load_frag(qrow + 0,  ksel);
  v16bf aq1 = load_frag(qrow + 32, ksel);
  const float sc = SCL[wave];

  // ---- Phase 1: QK^T per head -> scale -> diag assign -> pre-softmax 8x8 mix
  for (int c = 0; c < 8; ++c) {                 // 128-column chunks
    const int j0 = c * 128;
    for (int jt = 0; jt < 8; ++jt) {
      const int jb = j0 + jt * 16;
      const unsigned short* krow =
          kb + ((bb * H_ + wave) * N_ + (jb + mrow)) * DH_;
      v16bf b0 = load_frag(krow + 0,  ksel);
      v16bf b1 = load_frag(krow + 32, ksel);
      Acc acc;
#pragma unroll
      for (int i = 0; i < 8; ++i) acc.f[i] = 0.f;
      acc.v = __builtin_amdgcn_wmma_f32_16x16x32_bf16(false, aq0, false, b0,
                                                      (short)0, acc.v, false, false);
      acc.v = __builtin_amdgcn_wmma_f32_16x16x32_bf16(false, aq1, false, b1,
                                                      (short)0, acc.v, false, false);
      const int jg = jb + mrow;                 // this lane's column
#pragma unroll
      for (int v = 0; v < 8; ++v) {
        int ig = i0 + v + 8 * ksel;
        float d = acc.f[v] * sc;
        if (jg == ig) d = -1e-9f;               // assigned, not added (reference)
        RAW[((wave * 16 + v + 8 * ksel) << 7) + (jt * 16 + mrow)] = f2bf(d);
      }
    }
    __syncthreads();
    // mixed[g,i,j] = sum_h pre[h,g] * raw[h,i,j]
    for (int p = tid; p < 16 * 128; p += 256) {
      int i = p >> 7, j = p & 127;
      float r[8];
#pragma unroll
      for (int h = 0; h < 8; ++h) r[h] = bf2f(RAW[((h * 16 + i) << 7) + j]);
#pragma unroll
      for (int g = 0; g < 8; ++g) {
        float s = 0.f;
#pragma unroll
        for (int h = 0; h < 8; ++h) s += PRE[h * 8 + g] * r[h];
        S[((g * 16 + i) << 10) + (j0 + j)] = f2bf(s);
      }
    }
    __syncthreads();
  }

  // ---- Phase 2: row softmax of mixed head `wave`, fully in LDS
  for (int r = 0; r < 16; ++r) {
    unsigned short* row = S + ((wave * 16 + r) << 10);
    float m = -3.0e38f;
    for (int jc = lane; jc < N_; jc += 32) m = fmaxf(m, bf2f(row[jc]));
#pragma unroll
    for (int off = 16; off; off >>= 1) m = fmaxf(m, __shfl_xor(m, off, 32));
    float sum = 0.f;
    for (int jc = lane; jc < N_; jc += 32) {
      float e = __expf(bf2f(row[jc]) - m);
      row[jc] = f2bf(e);                        // unnormalized exp
      sum += e;
    }
#pragma unroll
    for (int off = 16; off; off >>= 1) sum += __shfl_xor(sum, off, 32);
    if (lane == 0) INVL[wave * 16 + r] = 1.f / sum;
  }
  __syncthreads();

  // ---- Phase 3: post-softmax mix folded into A-fragment build, then PV WMMA
  float coef[8];
#pragma unroll
  for (int h = 0; h < 8; ++h)
    coef[h] = POST[h * 8 + wave] * INVL[h * 16 + mrow];

  Acc out[4];
#pragma unroll
  for (int t = 0; t < 4; ++t)
#pragma unroll
    for (int i = 0; i < 8; ++i) out[t].f[i] = 0.f;

  for (int kb0 = 0; kb0 < N_; kb0 += 32) {
    Frag fa;                                    // attn_post[g] A-tile, built in regs
#pragma unroll
    for (int slot = 0; slot < 16; ++slot) {
      int j = kb0 + ksel * 8 + (slot & 7) + ((slot >= 8) ? 16 : 0);
      float s = 0.f;
#pragma unroll
      for (int h = 0; h < 8; ++h)
        s += coef[h] * bf2f(S[((h * 16 + mrow) << 10) + j]);
      fa.s[slot] = f2bf(s);
    }
#pragma unroll
    for (int dt = 0; dt < 4; ++dt) {
      const unsigned short* vrow =
          vTb + ((bb * H_ + wave) * DH_ + dt * 16 + mrow) * N_ + kb0;
      v16bf bv = load_frag(vrow, ksel);
      out[dt].v = __builtin_amdgcn_wmma_f32_16x16x32_bf16(false, fa.v, false, bv,
                                                          (short)0, out[dt].v,
                                                          false, false);
    }
  }

  // epilogue: ab[b, i, g*64 + d] bf16 for the output projection
#pragma unroll
  for (int dt = 0; dt < 4; ++dt)
#pragma unroll
    for (int v = 0; v < 8; ++v) {
      int ig = i0 + v + 8 * ksel;
      int d  = dt * 16 + mrow;
      ab[(bb * N_ + ig) * HD_ + wave * DH_ + d] = f2bf(out[dt].f[v]);
    }
}

// ------------------------------------------------------ output projection
// grid (B*N/16, 8): one wave computes a 16x64 tile, 4 wmma share one A frag.
__global__ void __launch_bounds__(32)
k_out(const unsigned short* __restrict__ ab,
      const unsigned short* __restrict__ woT,
      const float* __restrict__ bout,
      float* __restrict__ outp) {
  const int lane = threadIdx.x & 31;
  const int ksel = lane >> 4;
  const int mt = blockIdx.x;
  const int obase = blockIdx.y * 64;
  const unsigned short* arow  = ab  + (mt * 16 + (lane & 15)) * HD_;
  const unsigned short* brow0 = woT + (obase + (lane & 15)) * HD_;

  Acc acc[4];
#pragma unroll
  for (int j = 0; j < 4; ++j)
#pragma unroll
    for (int i = 0; i < 8; ++i) acc[j].f[i] = 0.f;

  for (int k = 0; k < HD_; k += 32) {
    __builtin_prefetch(arow + k + 64, 0, 1);
    v16bf a = load_frag(arow + k, ksel);
#pragma unroll
    for (int j = 0; j < 4; ++j) {
      __builtin_prefetch(brow0 + j * 16 * HD_ + k + 64, 0, 1);
      v16bf b = load_frag(brow0 + j * 16 * HD_ + k, ksel);
      acc[j].v = __builtin_amdgcn_wmma_f32_16x16x32_bf16(false, a, false, b,
                                                         (short)0, acc[j].v,
                                                         false, false);
    }
  }

#pragma unroll
  for (int j = 0; j < 4; ++j) {
    const int o = obase + j * 16 + (lane & 15);
    const float bias = bout[o];
#pragma unroll
    for (int v = 0; v < 8; ++v) {
      int m = mt * 16 + v + 8 * ksel;
      outp[m * DIM_ + o] = acc[j].f[v] + bias;
    }
  }
}

// --------------------------------------------------------------------- launch
extern "C" void kernel_launch(void* const* d_in, const int* in_sizes, int n_in,
                              void* d_out, int out_size, void* d_ws, size_t ws_size,
                              hipStream_t stream) {
  const float* x     = (const float*)d_in[0];
  const float* Wq    = (const float*)d_in[1];
  const float* Wkv   = (const float*)d_in[2];
  const float* scale = (const float*)d_in[3];
  const float* pre   = (const float*)d_in[4];
  const float* post  = (const float*)d_in[5];
  const float* Wout  = (const float*)d_in[6];
  const float* bout  = (const float*)d_in[7];
  float* outp = (float*)d_out;

  // workspace partition (bf16 elements), ~82 MB total
  unsigned short* ws   = (unsigned short*)d_ws;
  unsigned short* xb   = ws;                   // B*N*DIM      = 8388608
  unsigned short* wqT  = xb   + 8388608;       // HD*DIM       = 262144
  unsigned short* wkvT = wqT  + 262144;        // 2*HD*DIM     = 524288
  unsigned short* woT  = wkvT + 524288;        // DIM*HD       = 262144
  unsigned short* qb   = woT  + 262144;        // [b,h,n,d]    = 8388608
  unsigned short* kb   = qb   + 8388608;       // [b,h,n,d]
  unsigned short* vTb  = kb   + 8388608;       // [b,h,d,n]
  unsigned short* ab   = vTb  + 8388608;       // [b,n,h*d]

  k_cvt <<<(8388608 + 255) / 256, 256, 0, stream>>>(x, xb, 8388608);
  k_cvtT<<<(262144  + 255) / 256, 256, 0, stream>>>(Wq,   wqT,  512, 512);
  k_cvtT<<<(524288  + 255) / 256, 256, 0, stream>>>(Wkv,  wkvT, 512, 1024);
  k_cvtT<<<(262144  + 255) / 256, 256, 0, stream>>>(Wout, woT,  512, 512);

  k_proj<<<dim3(1024, 24), 32, 0, stream>>>(xb, wqT, wkvT, qb, kb, vTb);

  size_t smem = (size_t)(8 * 16 * 1024 + 8 * 16 * 128) * 2   // S + RAW (bf16)
              + (size_t)(8 * 16 + 64 + 64 + 8) * 4;          // INVL/PRE/POST/SCL
  k_attn<<<1024, 256, smem, stream>>>(qb, kb, vTb, scale, pre, post, ab);

  k_out<<<dim3(1024, 8), 32, 0, stream>>>(ab, woT, bout, outp);
}
